// StackLSTMParser_5196910428656
// MI455X (gfx1250) — matrix operand order repaired and checked by reading, not compiled, not run on hardware
//
#include <hip/hip_runtime.h>
#include <hip/hip_bf16.h>

typedef __attribute__((ext_vector_type(16))) __bf16 v16bf;
typedef __attribute__((ext_vector_type(8)))  unsigned int u32x8;
typedef __attribute__((ext_vector_type(8)))  float v8f;

namespace {
constexpr int NB     = 32;   // batch
constexpr int NSEQ   = 64;
constexpr int NSTEPS = 96;
constexpr int NH     = 256;  // HID == IN == WDIM == SDIM
constexpr int NS     = 24;   // stack size
constexpr int NACT   = 4;
constexpr int AD     = 64;   // action emb dim
}

__device__ __forceinline__ unsigned short f2bf(float x) {
  unsigned int u = __builtin_bit_cast(unsigned int, x);
  unsigned int r = u + 0x7fffu + ((u >> 16) & 1u);   // round-to-nearest-even
  return (unsigned short)(r >> 16);
}
__device__ __forceinline__ float sigm(float x) { return 1.0f / (1.0f + __expf(-x)); }
__device__ __forceinline__ int iclip(int x, int lo, int hi) { return x < lo ? lo : (x > hi ? hi : x); }

// Weights are pre-packed in per-lane fragment order (k_prep):
//   Bp[((nt*kblocks + kb)*32 + lane)*16 + j] = W^T[kb*32 + (j>>3)*16 + (lane>>4)*8 + (j&7)][nt*16 + (lane&15)]
// so a lane's whole 32-byte B fragment is contiguous -> two b128 loads.
// A (bf16, LDS, row-major, stride lda) fragment = two contiguous 16B runs per lane.
__device__ __forceinline__ void wave_gemm(const unsigned short* A, int lda,
                                          const unsigned short* __restrict__ Bp,
                                          float* __restrict__ D,
                                          int N, int K, int wave, int lane, int nwaves) {
  const int tilesN  = N >> 4;
  const int tiles   = 2 * tilesN;            // M = 32 -> 2 tile rows
  const int kblocks = K >> 5;
  const int half = lane >> 4, q = lane & 15;
  for (int t = wave; t < tiles; t += nwaves) {
    const int m0 = (t / tilesN) << 4;
    const int n0 = (t % tilesN) << 4;
    const unsigned short* __restrict__ bt = Bp + (size_t)(n0 >> 4) * kblocks * 512;
    const unsigned short* arow = A + (m0 + q) * lda + (half << 3);
    v8f acc = {};
    for (int kb = 0; kb < kblocks; ++kb) {
      const int k0 = kb << 5;
      const uint4 al = *(const uint4*)(arow + k0);        // K = k0+8h .. +7  (ds_load_b128)
      const uint4 ah = *(const uint4*)(arow + k0 + 16);   // K = k0+16+8h .. +7
      const uint4* bp = (const uint4*)(bt + ((size_t)(kb << 5) + lane) * 16);
      const uint4 bl = bp[0];                             // global_load_b128
      const uint4 bh = bp[1];
      u32x8 av, bv;
      av[0]=al.x; av[1]=al.y; av[2]=al.z; av[3]=al.w;
      av[4]=ah.x; av[5]=ah.y; av[6]=ah.z; av[7]=ah.w;
      bv[0]=bl.x; bv[1]=bl.y; bv[2]=bl.z; bv[3]=bl.w;
      bv[4]=bh.x; bv[5]=bh.y; bv[6]=bh.z; bv[7]=bh.w;
      acc = __builtin_amdgcn_wmma_f32_16x16x32_bf16(
          false, __builtin_bit_cast(v16bf, av),
          false, __builtin_bit_cast(v16bf, bv),
          (short)0, acc, false, false);
    }
#pragma unroll
    for (int r = 0; r < 8; ++r)
      D[(size_t)(m0 + r + (half << 3)) * N + n0 + q] = acc[r];
  }
}

// ---------------- weight prep: bf16 + fragment-order pack + fused bias ----------------
// decode i -> (k, n) of the packed layout for a K x N matrix
#define PACK_DECODE(i, kblocks)                         \
  const int j    = (i) & 15;                            \
  const int lane = ((i) >> 4) & 31;                     \
  const int blk  = (i) >> 9;                            \
  const int kb   = blk % (kblocks);                     \
  const int nt   = blk / (kblocks);                     \
  const int k    = (kb << 5) + ((j >> 3) << 4) + ((lane >> 4) << 3) + (j & 7);  \
  const int n    = (nt << 4) + (lane & 15);

__global__ void k_prep(const float* Wih_b, const float* Whh_b, const float* bih_b, const float* bhh_b,
                       const float* Wih_s, const float* Whh_s, const float* bih_s, const float* bhh_s,
                       const float* Wih_h, const float* Whh_h, const float* bih_h, const float* bhh_h,
                       const float* W_sum, const float* W_comp,
                       unsigned short* WTb, unsigned short* WTs, unsigned short* WTh,
                       unsigned short* WsumT, unsigned short* WcompT,
                       float* bias_b, float* bias_s, float* bias_h) {
  const int tid = blockIdx.x * blockDim.x + threadIdx.x;
  const int nt_ = gridDim.x * blockDim.x;
  for (int i = tid; i < 512 * 1024; i += nt_) {   // K=512 (x|h), N=1024, kblocks=16
    PACK_DECODE(i, 16)
    WTb[i] = f2bf(k < 256 ? Wih_b[n * 256 + k] : Whh_b[n * 256 + (k - 256)]);
    WTs[i] = f2bf(k < 256 ? Wih_s[n * 256 + k] : Whh_s[n * 256 + (k - 256)]);
  }
  for (int i = tid; i < 320 * 1024; i += nt_) {   // K=320 (a|h), N=1024, kblocks=10
    PACK_DECODE(i, 10)
    WTh[i] = f2bf(k < 64 ? Wih_h[n * 64 + k] : Whh_h[n * 256 + (k - 64)]);
  }
  for (int i = tid; i < 768 * 256; i += nt_) {    // K=768, N=256, kblocks=24
    PACK_DECODE(i, 24)
    WsumT[i] = f2bf(W_sum[n * 768 + k]);
  }
  for (int i = tid; i < 256 * 256; i += nt_) {    // K=256, N=256, kblocks=8
    PACK_DECODE(i, 8)
    WcompT[i] = f2bf(W_comp[n * 256 + k]);
  }
  for (int i = tid; i < 1024; i += nt_) {
    bias_b[i] = bih_b[i] + bhh_b[i];
    bias_s[i] = bih_s[i] + bhh_s[i];
    bias_h[i] = bih_h[i] + bhh_h[i];
  }
}

// ---------------- embedding gather + compose GEMM (2048 x 256 x 256) ----------------
__global__ void k_embed(const int* __restrict__ tokens, const float* __restrict__ emb,
                        const unsigned short* __restrict__ WcompT, const float* __restrict__ b_comp,
                        float* __restrict__ tbuf, unsigned short* __restrict__ xb) {
  const int lane = threadIdx.x & 31;
  const int wave = blockIdx.x * (blockDim.x >> 5) + (threadIdx.x >> 5);
  const int nw   = gridDim.x * (blockDim.x >> 5);
  const int half = lane >> 4, q = lane & 15;
  const int tilesN = NH >> 4;                       // 16
  const int tiles  = (NSEQ * NB / 16) * tilesN;     // 2048
  for (int t = wave; t < tiles; t += nw) {
    const int m0 = (t / tilesN) << 4;
    const int n0 = (t % tilesN) << 4;
    const int row = m0 + q;                         // r = s*NB + b
    const float* erow = emb + (size_t)tokens[row] * NH + (half << 3);
    const unsigned short* bt = WcompT + (size_t)(n0 >> 4) * 8 * 512;
    v8f acc = {};
    for (int kb = 0; kb < 8; ++kb) {
      const int k0 = kb << 5;
      const float4 f0 = *(const float4*)(erow + k0);       // global_load_b128 x4
      const float4 f1 = *(const float4*)(erow + k0 + 4);
      const float4 f2 = *(const float4*)(erow + k0 + 16);
      const float4 f3 = *(const float4*)(erow + k0 + 20);
      const uint4* bp = (const uint4*)(bt + ((size_t)(kb << 5) + lane) * 16);
      const uint4 bl = bp[0];
      const uint4 bh = bp[1];
      u32x8 av, bv;
      av[0] = (unsigned)f2bf(f0.x) | ((unsigned)f2bf(f0.y) << 16);
      av[1] = (unsigned)f2bf(f0.z) | ((unsigned)f2bf(f0.w) << 16);
      av[2] = (unsigned)f2bf(f1.x) | ((unsigned)f2bf(f1.y) << 16);
      av[3] = (unsigned)f2bf(f1.z) | ((unsigned)f2bf(f1.w) << 16);
      av[4] = (unsigned)f2bf(f2.x) | ((unsigned)f2bf(f2.y) << 16);
      av[5] = (unsigned)f2bf(f2.z) | ((unsigned)f2bf(f2.w) << 16);
      av[6] = (unsigned)f2bf(f3.x) | ((unsigned)f2bf(f3.y) << 16);
      av[7] = (unsigned)f2bf(f3.z) | ((unsigned)f2bf(f3.w) << 16);
      bv[0]=bl.x; bv[1]=bl.y; bv[2]=bl.z; bv[3]=bl.w;
      bv[4]=bh.x; bv[5]=bh.y; bv[6]=bh.z; bv[7]=bh.w;
      acc = __builtin_amdgcn_wmma_f32_16x16x32_bf16(
          false, __builtin_bit_cast(v16bf, av),
          false, __builtin_bit_cast(v16bf, bv),
          (short)0, acc, false, false);
    }
#pragma unroll
    for (int r = 0; r < 8; ++r) {
      const int m = m0 + r + (half << 3);
      const int s = m >> 5, b = m & 31, n = n0 + q;
      float v = acc[r] + b_comp[n];
      v = v > 0.f ? v : 0.f;                        // ReLU
      const size_t o = (size_t)(NSEQ - 1 - s) * NB * NH + b * NH + n;  // seq flip
      tbuf[o] = v;
      xb[o]   = f2bf(v);
    }
  }
}

// ---------------- persistent parser kernel ----------------
struct P {
  const unsigned short *WTb, *WTs, *WTh, *WsumT;
  const float *bias_b, *bias_s, *bias_h;
  const unsigned short *xb;                    // tok_rev bf16 (SEQ,B,H)
  float *tbuf, *buf, *Hs, *Cs, *tstk;          // mutable state
  float *g, *sinp, *sinp2, *nh, *nc, *ah, *ac, *hA, *cA;
  const int *mask, *actions;
  const float *h0, *c0, *root, *b_sum, *W_out, *b_out, *action_emb;
  float *out;
};

__device__ __forceinline__ void lstm_gate(const float* g, const float* bias,
                                          int b, int k, float cin, float& h2, float& c2) {
  const float gi = g[b * 1024 + k]       + bias[k];         // PyTorch order i,f,g,o
  const float gf = g[b * 1024 + 256 + k] + bias[256 + k];
  const float gg = g[b * 1024 + 512 + k] + bias[512 + k];
  const float go = g[b * 1024 + 768 + k] + bias[768 + k];
  c2 = sigm(gf) * cin + sigm(gi) * tanhf(gg);
  h2 = sigm(go) * tanhf(c2);
}

__launch_bounds__(1024)
__global__ void k_main(P p) {
  __shared__ __align__(16) unsigned short abuf[NB * 768];   // 48KB bf16 A staging
  __shared__ int spos[NB], bposA[NB], tposA[NB], tbposA[NB];
  __shared__ int sopA[NB], bopA[NB], aidA[NB], lenA[NB];
  __shared__ float logitsA[NB * NACT];

  const int tid = threadIdx.x, lane = tid & 31, wave = tid >> 5;
  const int NT = 1024;
  constexpr int SB = NB * NH;                  // 8192 f32 per (B,H) slab

  // ---- init ----
  if (tid < NB) {
    int L = 0;
    for (int s = 0; s < NSEQ; ++s) L += p.mask[s * NB + tid];
    lenA[tid] = L; bposA[tid] = L; tbposA[tid] = L; tposA[tid] = 0; spos[tid] = 0;
  }
  for (int e = tid; e < NS * SB; e += NT) { p.Hs[e] = 0.f; p.Cs[e] = 0.f; p.tstk[e] = 0.f; }
  for (int e = tid; e < SB; e += NT) {
    const int k = e & (NH - 1);
    p.hA[e] = p.h0[k]; p.cA[e] = p.c0[k];
    p.ah[e] = p.h0[k]; p.ac[e] = p.c0[k];
    p.Hs[e] = p.h0[k]; p.Cs[e] = p.c0[k];      // stack slot 0
  }
  __threadfence_block(); __syncthreads();
  for (int e = tid; e < SB; e += NT) {         // initial stack_input = top of tbuf
    const int b = e >> 8;
    const int tb = lenA[b];
    p.sinp[e] = (tb > 0) ? p.tbuf[(size_t)(tb - 1) * SB + e] : 0.f;
  }
  __threadfence_block(); __syncthreads();

  // ---- phase A: pre-buffer LSTM over reversed tokens ----
  for (int s = 0; s < NSEQ; ++s) {
    for (int e = tid; e < NB * 512; e += NT) {
      const int b = e >> 9, k = e & 511;
      abuf[b * 768 + k] = (k < 256) ? p.xb[(size_t)s * SB + b * NH + k]
                                    : f2bf(p.hA[b * NH + (k - 256)]);
    }
    __syncthreads();
    wave_gemm(abuf, 768, p.WTb, p.g, 1024, 512, wave, lane, 32);
    __threadfence_block(); __syncthreads();
    for (int e = tid; e < SB; e += NT) {
      const int b = e >> 8, k = e & 255;
      float h2, c2;
      lstm_gate(p.g, p.bias_b, b, k, p.cA[e], h2, c2);
      p.cA[e] = c2; p.hA[e] = h2;
      p.buf[(size_t)s * SB + e] = h2;          // buffer_hiddens
    }
    __threadfence_block(); __syncthreads();
  }

  // ---- root push onto stack LSTM ----
  for (int e = tid; e < NB * 512; e += NT) {
    const int b = e >> 9, k = e & 511;
    abuf[b * 768 + k] = (k < 256) ? f2bf(p.root[k]) : f2bf(p.h0[k - 256]);
  }
  __syncthreads();
  wave_gemm(abuf, 768, p.WTs, p.g, 1024, 512, wave, lane, 32);
  __threadfence_block(); __syncthreads();
  for (int e = tid; e < SB; e += NT) {
    const int b = e >> 8, k = e & 255;
    float h2, c2;
    lstm_gate(p.g, p.bias_s, b, k, p.c0[k], h2, c2);
    p.Hs[(size_t)1 * SB + e] = h2; p.Cs[(size_t)1 * SB + e] = c2;
  }
  if (tid < NB) spos[tid] = 1;
  __threadfence_block(); __syncthreads();

  // ---- phase B: 96 parser steps ----
  for (int t = 0; t < NSTEPS; ++t) {
    // summary input = [stack_head | bhead | ah]  (32 x 768)
    for (int e = tid; e < NB * 768; e += NT) {
      const int b = e / 768, k = e - b * 768;
      float v;
      if (k < 256) v = p.Hs[(size_t)spos[b] * SB + b * NH + k];
      else if (k < 512) {
        const int bp = bposA[b], kk = k - 256;
        v = (bp > 0) ? p.buf[(size_t)(bp - 1) * SB + b * NH + kk] : p.h0[kk];
      } else v = p.ah[b * NH + (k - 512)];
      abuf[b * 768 + k] = f2bf(v);
    }
    __syncthreads();
    wave_gemm(abuf, 768, p.WsumT, p.g, 256, 768, wave, lane, 32);
    __threadfence_block(); __syncthreads();
    for (int e = tid; e < NB * 256; e += NT) { // bias + ReLU in place
      float v = p.g[e] + p.b_sum[e & 255];
      p.g[e] = v > 0.f ? v : 0.f;
    }
    __threadfence_block(); __syncthreads();
    if (tid < NB * NACT) {                     // logits (tiny: 32x4 of K=256)
      const int b = tid >> 2, a = tid & 3;
      float s = p.b_out[a];
      const float4* gr = (const float4*)(p.g + b * 256);
      const float4* wr = (const float4*)(p.W_out + a * 256);
      for (int k = 0; k < 64; ++k) {
        const float4 gv = gr[k], wv = wr[k];
        s += gv.x * wv.x + gv.y * wv.y + gv.z * wv.z + gv.w * wv.w;
      }
      logitsA[tid] = s;
    }
    __syncthreads();
    if (tid < NB) {                            // log_softmax + decode oracle action
      const int b = tid;
      const float l0 = logitsA[b*4], l1 = logitsA[b*4+1], l2 = logitsA[b*4+2], l3 = logitsA[b*4+3];
      const float m = fmaxf(fmaxf(l0, l1), fmaxf(l2, l3));
      const float ls = m + __logf(__expf(l0-m)+__expf(l1-m)+__expf(l2-m)+__expf(l3-m));
      float* o = p.out + (size_t)t * NB * NACT + b * NACT;
      o[0] = l0 - ls; o[1] = l1 - ls; o[2] = l2 - ls; o[3] = l3 - ls;
      const int a = p.actions[t * NB + b];
      aidA[b] = a;
      sopA[b] = (a == 0 || a == 3) ? 1 : -1;   // STACK_MAP
      bopA[b] = (a == 1 || a == 2) ? 0 : -1;   // BUF_MAP
    }
    __syncthreads();

    // stack LSTM: x = stack_input, h = Hs[spos]
    for (int e = tid; e < NB * 512; e += NT) {
      const int b = e >> 9, k = e & 511;
      const float v = (k < 256) ? p.sinp[b * NH + k]
                                : p.Hs[(size_t)spos[b] * SB + b * NH + (k - 256)];
      abuf[b * 768 + k] = f2bf(v);
    }
    __syncthreads();
    wave_gemm(abuf, 768, p.WTs, p.g, 1024, 512, wave, lane, 32);
    __threadfence_block(); __syncthreads();
    for (int e = tid; e < SB; e += NT) {
      const int b = e >> 8, k = e & 255;
      float h2, c2;
      lstm_gate(p.g, p.bias_s, b, k, p.Cs[(size_t)spos[b] * SB + e], h2, c2);
      p.nh[e] = h2; p.nc[e] = c2;
    }
    __threadfence_block(); __syncthreads();

    // pointer machinery: stack push/pop, buf / tstk / tbuf state-stacks
    for (int e = tid; e < SB; e += NT) {
      const int b = e >> 8;
      const int so = sopA[b], bo = bopA[b];
      const int nsp = iclip(spos[b] + so, 0, NS - 1);
      float nsh;
      if (so == 1) {
        nsh = p.nh[e];
        p.Hs[(size_t)nsp * SB + e] = nsh;
        p.Cs[(size_t)nsp * SB + e] = p.nc[e];
      } else nsh = p.Hs[(size_t)nsp * SB + e];
      const int nbp = iclip(bposA[b] + bo, 0, NSEQ);
      const int wi  = iclip(nbp - 1, 0, NSEQ - 1);
      if (bo == 1) p.buf[(size_t)wi * SB + e] = nsh;
      const float si = p.sinp[e];
      const int ntp = iclip(tposA[b] + so, 0, NS);
      const int wi2 = iclip(ntp - 1, 0, NS - 1);
      if (so == 1) p.tstk[(size_t)wi2 * SB + e] = si;
      const int ntb = iclip(tbposA[b] + bo, 0, NSEQ);
      const int wi3 = iclip(ntb - 1, 0, NSEQ - 1);
      if (bo == 1) p.tbuf[(size_t)wi3 * SB + e] = si;
      p.sinp2[e] = (ntb > 0) ? ((bo == 1) ? si : p.tbuf[(size_t)wi3 * SB + e]) : 0.f;
    }
    __threadfence_block(); __syncthreads();
    if (tid < NB) {
      const int b = tid;
      spos[b]   = iclip(spos[b] + sopA[b], 0, NS - 1);
      bposA[b]  = iclip(bposA[b] + bopA[b], 0, NSEQ);
      tposA[b]  = iclip(tposA[b] + sopA[b], 0, NS);
      tbposA[b] = iclip(tbposA[b] + bopA[b], 0, NSEQ);
    }
    for (int e = tid; e < SB; e += NT) p.sinp[e] = p.sinp2[e];
    __threadfence_block(); __syncthreads();

    // action LSTM: x = action_emb[a] (64), h = ah (256) -> K = 320
    for (int e = tid; e < NB * 320; e += NT) {
      const int b = e / 320, k = e - b * 320;
      const float v = (k < 64) ? p.action_emb[aidA[b] * AD + k] : p.ah[b * NH + (k - 64)];
      abuf[b * 768 + k] = f2bf(v);
    }
    __syncthreads();
    wave_gemm(abuf, 768, p.WTh, p.g, 1024, 320, wave, lane, 32);
    __threadfence_block(); __syncthreads();
    for (int e = tid; e < SB; e += NT) {
      const int b = e >> 8, k = e & 255;
      float h2, c2;
      lstm_gate(p.g, p.bias_h, b, k, p.ac[e], h2, c2);
      p.ac[e] = c2; p.ah[e] = h2;
    }
    __threadfence_block(); __syncthreads();
  }
}

extern "C" void kernel_launch(void* const* d_in, const int* in_sizes, int n_in,
                              void* d_out, int out_size, void* d_ws, size_t ws_size,
                              hipStream_t stream) {
  (void)in_sizes; (void)n_in; (void)out_size;
  const int*   tokens     = (const int*)  d_in[0];
  const int*   mask       = (const int*)  d_in[1];
  const int*   actions    = (const int*)  d_in[2];
  const float* emb_word   = (const float*)d_in[3];
  const float* W_comp     = (const float*)d_in[4];
  const float* b_comp     = (const float*)d_in[5];
  const float* root       = (const float*)d_in[6];
  const float* h0         = (const float*)d_in[7];
  const float* c0         = (const float*)d_in[8];
  const float* Wih_b      = (const float*)d_in[9];
  const float* Whh_b      = (const float*)d_in[10];
  const float* bih_b      = (const float*)d_in[11];
  const float* bhh_b      = (const float*)d_in[12];
  const float* Wih_s      = (const float*)d_in[13];
  const float* Whh_s      = (const float*)d_in[14];
  const float* bih_s      = (const float*)d_in[15];
  const float* bhh_s      = (const float*)d_in[16];
  const float* Wih_h      = (const float*)d_in[17];
  const float* Whh_h      = (const float*)d_in[18];
  const float* bih_h      = (const float*)d_in[19];
  const float* bhh_h      = (const float*)d_in[20];
  const float* action_emb = (const float*)d_in[21];
  const float* W_sum      = (const float*)d_in[22];
  const float* b_sum      = (const float*)d_in[23];
  const float* W_out      = (const float*)d_in[24];
  const float* b_out      = (const float*)d_in[25];

  char* w = (char*)d_ws;
  auto alloc = [&](size_t bytes) -> void* {
    void* q = (void*)w;
    w += (bytes + 255) & ~(size_t)255;
    return q;
  };
  unsigned short* WTb    = (unsigned short*)alloc(512 * 1024 * 2);
  unsigned short* WTs    = (unsigned short*)alloc(512 * 1024 * 2);
  unsigned short* WTh    = (unsigned short*)alloc(320 * 1024 * 2);
  unsigned short* WsumT  = (unsigned short*)alloc(768 * 256 * 2);
  unsigned short* WcompT = (unsigned short*)alloc(256 * 256 * 2);
  unsigned short* xb     = (unsigned short*)alloc((size_t)NSEQ * NB * NH * 2);
  float* bias_b = (float*)alloc(1024 * 4);
  float* bias_s = (float*)alloc(1024 * 4);
  float* bias_h = (float*)alloc(1024 * 4);
  float* tbuf   = (float*)alloc((size_t)NSEQ * NB * NH * 4);
  float* buf    = (float*)alloc((size_t)NSEQ * NB * NH * 4);
  float* Hs     = (float*)alloc((size_t)NS * NB * NH * 4);
  float* Cs     = (float*)alloc((size_t)NS * NB * NH * 4);
  float* tstk   = (float*)alloc((size_t)NS * NB * NH * 4);
  float* g      = (float*)alloc((size_t)NB * 1024 * 4);
  float* sinp   = (float*)alloc((size_t)NB * NH * 4);
  float* sinp2  = (float*)alloc((size_t)NB * NH * 4);
  float* nh     = (float*)alloc((size_t)NB * NH * 4);
  float* nc     = (float*)alloc((size_t)NB * NH * 4);
  float* ah     = (float*)alloc((size_t)NB * NH * 4);
  float* ac     = (float*)alloc((size_t)NB * NH * 4);
  float* hA     = (float*)alloc((size_t)NB * NH * 4);
  float* cA     = (float*)alloc((size_t)NB * NH * 4);
  if ((size_t)(w - (char*)d_ws) > ws_size) return;  // scratch too small

  k_prep<<<256, 256, 0, stream>>>(Wih_b, Whh_b, bih_b, bhh_b,
                                  Wih_s, Whh_s, bih_s, bhh_s,
                                  Wih_h, Whh_h, bih_h, bhh_h,
                                  W_sum, W_comp,
                                  WTb, WTs, WTh, WsumT, WcompT,
                                  bias_b, bias_s, bias_h);
  k_embed<<<32, 256, 0, stream>>>(tokens, emb_word, WcompT, b_comp, tbuf, xb);

  P p;
  p.WTb = WTb; p.WTs = WTs; p.WTh = WTh; p.WsumT = WsumT;
  p.bias_b = bias_b; p.bias_s = bias_s; p.bias_h = bias_h;
  p.xb = xb; p.tbuf = tbuf; p.buf = buf; p.Hs = Hs; p.Cs = Cs; p.tstk = tstk;
  p.g = g; p.sinp = sinp; p.sinp2 = sinp2; p.nh = nh; p.nc = nc;
  p.ah = ah; p.ac = ac; p.hA = hA; p.cA = cA;
  p.mask = mask; p.actions = actions;
  p.h0 = h0; p.c0 = c0; p.root = root; p.b_sum = b_sum;
  p.W_out = W_out; p.b_out = b_out; p.action_emb = action_emb;
  p.out = (float*)d_out;
  k_main<<<1, 1024, 0, stream>>>(p);
}
